// MSNN_60241211294185
// MI455X (gfx1250) — matrix operand order repaired and checked by reading, not compiled, not run on hardware
//
#include <hip/hip_runtime.h>

#define BATCH 4096
#define NI 1024
#define NH 512
#define NO 256
#define TSTEPS 50
#define BETA 0.95f
#define THRESH 1.0f
#define BM 32               // batch rows per block in the time loop
#define SPK_LD (NH + 8)     // padded bf16 row stride: 1040B -> 4-bank skew/row
#define W2_LD (NH + 8)      // same padding for LDS-resident W2h

typedef float  v2f   __attribute__((ext_vector_type(2)));
typedef float  v8f   __attribute__((ext_vector_type(8)));
typedef __bf16 v16bf __attribute__((ext_vector_type(16)));
typedef unsigned int u32x4 __attribute__((ext_vector_type(4)));
typedef int          i32x4 __attribute__((ext_vector_type(4)));
typedef int          i32x8 __attribute__((ext_vector_type(8)));

union BF16x16 { v16bf v; uint4 q[2]; };

// ---------------------------------------------------------------------------
// Prep: split W2 (f32) into hi/lo bf16 so spk1@W2^T can run on bf16 WMMA with
// ~fp32 effective precision (spikes are exactly representable in bf16).
// ---------------------------------------------------------------------------
__global__ void snn_split_w2(const float* __restrict__ W2,
                             __bf16* __restrict__ W2h, __bf16* __restrict__ W2l) {
    int i = blockIdx.x * blockDim.x + threadIdx.x;
    if (i < NO * NH) {
        float w = W2[i];
        __bf16 h = (__bf16)w;
        float  r = w - (float)h;
        W2h[i] = h;
        W2l[i] = (__bf16)r;
    }
}

// ---------------------------------------------------------------------------
// GEMM1: cur1 = x @ W1^T + b1   (exact fp32 via v_wmma_f32_16x16x4_f32)
// ---------------------------------------------------------------------------
__global__ __launch_bounds__(256) void snn_gemm1(const float* __restrict__ x,
                                                 const float* __restrict__ W1,
                                                 const float* __restrict__ b1,
                                                 float* __restrict__ cur1) {
    const int lane = threadIdx.x & 31;
    const int wave = threadIdx.x >> 5;
    const int lo   = lane & 15;
    const int half = lane >> 4;
    const int r0 = blockIdx.x * 32 + (wave >> 2) * 16;
    const int n0 = blockIdx.y * 256 + (wave & 3) * 64;

    const float* arow = x + (size_t)(r0 + lo) * NI;
    v8f acc[4] = {};

    for (int k = 0; k < NI; k += 4) {
        v2f a = *(const v2f*)(arow + k + 2 * half);
#pragma unroll
        for (int nt = 0; nt < 4; ++nt) {
            v2f b = *(const v2f*)(W1 + (size_t)(n0 + nt * 16 + lo) * NI + k + 2 * half);
            acc[nt] = __builtin_amdgcn_wmma_f32_16x16x4_f32(
                false, a, false, b, (short)0, acc[nt], false, false);
        }
    }

#pragma unroll
    for (int nt = 0; nt < 4; ++nt) {
        int col = n0 + nt * 16 + lo;
        float bias = b1[col];
#pragma unroll
        for (int v = 0; v < 8; ++v) {
            int row = r0 + v + 8 * half;
            cur1[(size_t)row * NH + col] = acc[nt][v] + bias;
        }
    }
}

// ---------------------------------------------------------------------------
// Time loop: 128 blocks x 512 threads (16 waves), 32 batch rows per block.
//  - mem1/cur1: REGISTERS (thread t owns column t for all 32 rows)
//  - spk1:      LDS (32.5 KB, padded)
//  - W2h:       LDS-resident, staged once via the Tensor Data Mover (TDM):
//               2D D# with pad_interval=256dw/pad_amount=4dw reproduces the
//               520-bf16 bank-skewed row stride in hardware.
//  - W2l:       streamed from L2 each step (1.6 GB machine-wide total)
//  - mem2:      registers in WMMA C/D lane layout
// ---------------------------------------------------------------------------
__global__ __launch_bounds__(512) void snn_timeloop(const float* __restrict__ cur1g,
                                                    const __bf16* __restrict__ W2h,
                                                    const __bf16* __restrict__ W2l,
                                                    const float* __restrict__ b2,
                                                    float* __restrict__ out) {
    extern __shared__ char smem_raw[];
    __bf16* s_w2h  = (__bf16*)smem_raw;                    // 256 * 520 bf16 = 260 KB
    __bf16* s_spk1 = s_w2h + NO * W2_LD;                   // 32 * 520 bf16  = 32.5 KB

    const int tid    = threadIdx.x;
    const int lane   = tid & 31;
    const int wave   = tid >> 5;            // 16 waves
    const int lo     = lane & 15;
    const int half   = lane >> 4;
    const int rg     = wave >> 3;           // row group: rows rg*16..+15
    const int cg     = wave & 7;            // col group: cols cg*32..+31
    const int n0     = cg * 32;
    const int rbase  = rg * 16;
    const int rowblk = blockIdx.x * BM;

    // ---- Stage W2h into LDS (once) via TDM with hardware row padding ----
#if __has_builtin(__builtin_amdgcn_tensor_load_to_lds)
    if (wave == 0) {
        unsigned long long ga = (unsigned long long)(uintptr_t)W2h;
        u32x4 g0;
        g0[0] = 1u;                                     // count=1 user descriptor
        g0[1] = 0u;                                     // lds_addr: dynamic-LDS base
        g0[2] = (unsigned)(ga & 0xFFFFFFFFu);           // global_addr[31:0]
        g0[3] = (unsigned)((ga >> 32) & 0x01FFFFFFu)    // global_addr[56:32]
              | (2u << 30);                             // type = 2 ("image")
        i32x8 g1;
        g1[0] = (1 << 16)                               // data_size = 2 bytes
              | (1 << 20)                               // pad_enable
              | (7 << 22)                               // pad_interval: 256 DWORDs
              | (3 << 25);                              // pad_amount:   4 DWORDs (16 B)
        g1[1] = (int)(512u << 16);                      // tensor_dim0 = 512
        g1[2] = (int)(256u << 16);                      // tensor_dim1 = 256
        g1[3] = (int)(512u << 16);                      // tile_dim0   = 512
        g1[4] = 256;                                    // tile_dim1 = 256, tile_dim2 = 0
        g1[5] = 512;                                    // tensor_dim0_stride = 512
        g1[6] = 0;
        g1[7] = 0;
        i32x4 z4 = {0, 0, 0, 0};
#if defined(__clang_major__) && (__clang_major__ >= 23)
        i32x8 z8 = {0, 0, 0, 0, 0, 0, 0, 0};
        __builtin_amdgcn_tensor_load_to_lds(g0, g1, z4, z4, z8, 0);
#else
        __builtin_amdgcn_tensor_load_to_lds(g0, g1, z4, z4, 0);
#endif
        __builtin_amdgcn_s_wait_tensorcnt(0);
    }
#else
    {
        const uint4* src = (const uint4*)W2h;           // 16 B = 8 bf16 chunks
        for (int i = tid; i < NO * NH / 8; i += 512) {
            int r = i >> 6;
            int c = i & 63;
            *(uint4*)(s_w2h + r * W2_LD + c * 8) = src[i];
        }
    }
#endif

    // ---- Per-thread persistent state: column `tid` of rows 0..31 ----
    float mem1[BM], cur1[BM];
#pragma unroll
    for (int r = 0; r < BM; ++r) {
        mem1[r] = 0.0f;
        cur1[r] = cur1g[(size_t)(rowblk + r) * NH + tid];
    }

    float mem2[2][8];
#pragma unroll
    for (int nt = 0; nt < 2; ++nt)
#pragma unroll
        for (int v = 0; v < 8; ++v) mem2[nt][v] = 0.0f;

    const float bias[2] = { b2[n0 + lo], b2[n0 + 16 + lo] };

    float* outspk = out;
    float* outmem = out + (size_t)TSTEPS * BATCH * NO;

    __syncthreads();   // W2h staged (TDM complete + visible)

    for (int t = 0; t < TSTEPS; ++t) {
        // ---- Layer-1 leaky update + spike (thread owns one column) ----
#pragma unroll
        for (int r = 0; r < BM; ++r) {
            float m   = mem1[r];
            float rst = (m > THRESH) ? THRESH : 0.0f;    // reset uses PREVIOUS mem
            m = BETA * m + cur1[r] - rst;
            mem1[r] = m;
            s_spk1[r * SPK_LD + tid] = (m > THRESH) ? (__bf16)1.0f : (__bf16)0.0f;
        }
        __syncthreads();

        // ---- GEMM2: 16x32 tile per wave, K = 512 ----
        v8f acc[2] = {};
        const __bf16* arow = s_spk1 + (rbase + lo) * SPK_LD;
        for (int ks = 0; ks < NH; ks += 32) {
            BF16x16 a;
            a.q[0] = *(const uint4*)(arow + ks + 16 * half);
            a.q[1] = *(const uint4*)(arow + ks + 16 * half + 8);
#pragma unroll
            for (int nt = 0; nt < 2; ++nt) {
                const int brow = n0 + nt * 16 + lo;
                BF16x16 bh;
                const __bf16* hsrc = s_w2h + brow * W2_LD + ks + 16 * half;
                bh.q[0] = *(const uint4*)(hsrc);
                bh.q[1] = *(const uint4*)(hsrc + 8);
                BF16x16 bl;
                const __bf16* lsrc = W2l + (size_t)brow * NH + ks + 16 * half;
                bl.q[0] = *(const uint4*)(lsrc);
                bl.q[1] = *(const uint4*)(lsrc + 8);
                acc[nt] = __builtin_amdgcn_wmma_f32_16x16x32_bf16(
                    false, a.v, false, bh.v, (short)0, acc[nt], false, false);
                acc[nt] = __builtin_amdgcn_wmma_f32_16x16x32_bf16(
                    false, a.v, false, bl.v, (short)0, acc[nt], false, false);
            }
        }

        // ---- Layer-2 leaky update + spike; record outputs ----
#pragma unroll
        for (int nt = 0; nt < 2; ++nt) {
            int col = n0 + nt * 16 + lo;
#pragma unroll
            for (int v = 0; v < 8; ++v) {
                float m   = mem2[nt][v];
                float rst = (m > THRESH) ? THRESH : 0.0f;
                m = BETA * m + (acc[nt][v] + bias[nt]) - rst;
                mem2[nt][v] = m;
                int grow = rowblk + rbase + v + 8 * half;
                size_t idx = ((size_t)t * BATCH + grow) * NO + col;
                outspk[idx] = (m > THRESH) ? 1.0f : 0.0f;
                outmem[idx] = m;
            }
        }
        __syncthreads();   // spk1 fully consumed before next step rewrites it
    }
}

// ---------------------------------------------------------------------------
extern "C" void kernel_launch(void* const* d_in, const int* in_sizes, int n_in,
                              void* d_out, int out_size, void* d_ws, size_t ws_size,
                              hipStream_t stream) {
    const float* x  = (const float*)d_in[0];
    const float* W1 = (const float*)d_in[1];
    const float* b1 = (const float*)d_in[2];
    const float* W2 = (const float*)d_in[3];
    const float* b2 = (const float*)d_in[4];
    float* out = (float*)d_out;

    char* ws = (char*)d_ws;
    float*  cur1 = (float*)ws;                                   // 8 MB
    __bf16* W2h  = (__bf16*)(ws + (size_t)BATCH * NH * sizeof(float));
    __bf16* W2l  = W2h + (size_t)NO * NH;                        // 256 KB each

    (void)in_sizes; (void)n_in; (void)out_size; (void)ws_size;

    const size_t smem = (size_t)NO * W2_LD * sizeof(__bf16)
                      + (size_t)BM * SPK_LD * sizeof(__bf16);
    hipFuncSetAttribute((const void*)snn_timeloop,
                        hipFuncAttributeMaxDynamicSharedMemorySize, (int)smem);

    snn_split_w2<<<(NO * NH + 255) / 256, 256, 0, stream>>>(W2, W2h, W2l);

    dim3 g1(BATCH / 32, NH / 256);
    snn_gemm1<<<g1, 256, 0, stream>>>(x, W1, b1, cur1);

    snn_timeloop<<<BATCH / BM, 512, smem, stream>>>(cur1, W2h, W2l, b2, out);
}
